// FactorizedDenseAttention_60447369724632
// MI455X (gfx1250) — compile-verified, matrix operand order checked by reading, not run on hardware
//
#include <hip/hip_runtime.h>
#include <hip/hip_bf16.h>
#include <math.h>

// ---------------------------------------------------------------------------
// FactorizedDenseAttention on MI455X (gfx1250)
//
//   A = (q@Wa + ba) * 32          [H,S,32]  (f16, repeat factor folded in)
//   C[s,i] = b[s,2i] + b[s,2i+1], b = q@Wb + bb    [H,S,32] (f16)
//   scores = A @ C^T   (K=32  -> one v_wmma_f32_16x16x32_f16 per 16x16 tile)
//   attn   = softmax(scores)      (full 16x2048 strip resident in 128KB LDS)
//   out    = attn @ V  (K=2048 -> 64 accumulating WMMAs per 16x16 tile;
//                       V tiles staged into LDS by the Tensor Data Mover,
//                       double buffered, tracked with TENSORcnt)
// ---------------------------------------------------------------------------

typedef __attribute__((ext_vector_type(16))) _Float16 v16h;
typedef __attribute__((ext_vector_type(8)))  float    v8f;

#define NHEAD 16
#define SEQ   2048
#define DK    64

// LDS layout (floats) for the attention kernel
#define SC_FLOATS   (16 * SEQ)              // 32768 : score strip [16][2048]
#define RED_FLOATS  256                     // reduction scratch
#define VST_FLOATS  (32 * 16)               // 512 : one V sub-tile (32 rows x 16 cols)
#define VST_BASE    (SC_FLOATS + RED_FLOATS)
#define LDS_FLOATS  (VST_BASE + 8 * VST_FLOATS)   // 4 waves x 2 buffers

// ---------------------------------------------------------------------------
// Kernel 1: per-row projections  ->  A (f16, x32) and C (f16)
// ---------------------------------------------------------------------------
__global__ __launch_bounds__(256)
void fda_precompute(const float* __restrict__ q,
                    const float* __restrict__ Wa, const float* __restrict__ ba,
                    const float* __restrict__ Wb, const float* __restrict__ bb,
                    _Float16* __restrict__ Af, _Float16* __restrict__ Cf)
{
    __shared__ float sWa[DK * 32];
    __shared__ float sWb[DK * 64];
    __shared__ float sba[32];
    __shared__ float sbb[64];

    const int tid = threadIdx.x;
    for (int i = tid; i < DK * 32; i += 256) sWa[i] = Wa[i];
    for (int i = tid; i < DK * 64; i += 256) sWb[i] = Wb[i];
    if (tid < 32) sba[tid] = ba[tid];
    if (tid < 64) sbb[tid] = bb[tid];
    __syncthreads();

    const int r = blockIdx.x * 256 + tid;        // 0 .. H*S-1
    const float* __restrict__ qrow = q + (size_t)r * DK;

    float qr[DK];
    #pragma unroll
    for (int i = 0; i < DK / 4; ++i) {
        const float4 f = ((const float4*)qrow)[i];
        qr[4*i+0] = f.x; qr[4*i+1] = f.y; qr[4*i+2] = f.z; qr[4*i+3] = f.w;
    }

    _Float16* __restrict__ arow = Af + (size_t)r * 32;
    #pragma unroll 4
    for (int j = 0; j < 32; ++j) {
        float acc = 0.f;
        #pragma unroll
        for (int k = 0; k < DK; ++k) acc = fmaf(qr[k], sWa[k * 32 + j], acc);
        arow[j] = (_Float16)((acc + sba[j]) * 32.0f);   // fold repeat factor
    }

    _Float16* __restrict__ crow = Cf + (size_t)r * 32;
    #pragma unroll 4
    for (int i = 0; i < 32; ++i) {
        float a0 = 0.f, a1 = 0.f;
        #pragma unroll
        for (int k = 0; k < DK; ++k) {
            const float qk = qr[k];
            a0 = fmaf(qk, sWb[k * 64 + 2 * i + 0], a0);
            a1 = fmaf(qk, sWb[k * 64 + 2 * i + 1], a1);
        }
        crow[i] = (_Float16)(a0 + a1 + sbb[2 * i] + sbb[2 * i + 1]);
    }
}

// Gather one 16x32 f16 operand fragment (ISA 16-bit A layout):
// lanes 0-15 hold K {0..7,16..23}, lanes 16-31 hold K {8..15,24..31}.
__device__ __forceinline__ v16h load_frag_f16(const _Float16* __restrict__ row, int hi)
{
    v16h f;
    #pragma unroll
    for (int i = 0; i < 8; ++i) {
        f[i]     = row[hi * 8 + i];
        f[8 + i] = row[16 + hi * 8 + i];
    }
    return f;
}

// ---------------------------------------------------------------------------
// Tensor Data Mover: DMA one 32x16 f32 sub-tile of V into LDS.
//   - tensor: [2048][64] f32 row-major (this head's V), dim0 stride = 64
//   - tile:   tile_dim0 = 16 cols (contiguous), tile_dim1 = 32 rows
//   - LDS:    packed contiguous 32x16 floats at lds_byte_off
// Descriptor per CDNA5 ISA; tracked with TENSORcnt.
// This toolchain exposes the 6-arg builtin:
//   (uint32x4 g0, int32x8 g1, int32x4 g2, int32x4 g3, int32x8, i32 cpol)
// ---------------------------------------------------------------------------
#if __has_builtin(__builtin_amdgcn_tensor_load_to_lds)
#define FDA_HAS_TDM 1
typedef __attribute__((ext_vector_type(4))) unsigned int v4u;
typedef __attribute__((ext_vector_type(8))) int v8i_t;
typedef __attribute__((ext_vector_type(4))) int v4i_t;

__device__ __forceinline__ void tdm_load_v_tile(const float* gsrc, unsigned lds_byte_off)
{
    const unsigned long long ga = (unsigned long long)(uintptr_t)gsrc;
    v4u g0;
    g0[0] = 1u;                                   // count=1 (valid user D#)
    g0[1] = lds_byte_off;                         // lds_addr (bytes)
    g0[2] = (unsigned)(ga & 0xffffffffu);         // global_addr[31:0]
    g0[3] = (unsigned)(ga >> 32) | (2u << 30);    // global_addr[56:32] | type=2
    v8i_t g1;
    g1[0] = 0x20000;                              // data_size=2 (4 bytes)
    g1[1] = (int)(64u << 16);                     // tensor_dim0 = 64  (lo16)
    g1[2] = (int)(2048u << 16);                   // tensor_dim1 = 2048 (lo16)
    g1[3] = (int)(16u << 16);                     // tile_dim0 = 16
    g1[4] = 32;                                   // tile_dim1 = 32, tile_dim2 = 0
    g1[5] = 64;                                   // tensor_dim0_stride = 64
    g1[6] = 0;
    g1[7] = 0;
    const v4i_t z4 = {0, 0, 0, 0};                // groups 2/3: unused (2D tensor)
    const v8i_t z8 = {0, 0, 0, 0, 0, 0, 0, 0};
    __builtin_amdgcn_tensor_load_to_lds(g0, g1, z4, z4, z8, 0);
}
#else
#define FDA_HAS_TDM 0
#endif

// ---------------------------------------------------------------------------
// Kernel 2: fused scores -> softmax -> attn writeback -> attn@V
// grid = H * (S/16) blocks, 256 threads (8 waves), ~145KB dynamic LDS
// ---------------------------------------------------------------------------
__global__ __launch_bounds__(256)
void fda_attention(const _Float16* __restrict__ Af,
                   const _Float16* __restrict__ Cf,
                   const float* __restrict__ v,
                   float* __restrict__ outp,      // [H,S,64]
                   float* __restrict__ attnp)     // [H,S,S]
{
    extern __shared__ float sm[];
    float* sc  = sm;                  // scores strip [16][2048]
    float* red = sm + SC_FLOATS;      // reduction scratch [256]

    const int h    = blockIdx.x >> 7;          // S/16 == 128 tiles per head
    const int q0   = (blockIdx.x & 127) * 16;
    const int tid  = threadIdx.x;
    const int lane = tid & 31;
    const int wave = tid >> 5;
    const int mrow = lane & 15;
    const int hi   = lane >> 4;

    // ---- Phase 1: scores = A @ C^T via WMMA (K=32, one op per tile) ----
    const v16h afrag =
        load_frag_f16(Af + ((size_t)(h * SEQ + q0 + mrow)) * 32, hi);

    for (int t = wave; t < SEQ / 16; t += 8) {
        const v16h bfrag =
            load_frag_f16(Cf + ((size_t)(h * SEQ + t * 16 + mrow)) * 32, hi);
        v8f acc = {};
        acc = __builtin_amdgcn_wmma_f32_16x16x32_f16(
                  false, afrag, false, bfrag, (short)0, acc, false, false);
        #pragma unroll
        for (int r = 0; r < 8; ++r)
            sc[(r + 8 * hi) * SEQ + t * 16 + mrow] = acc[r];
    }
    __syncthreads();

    // ---- Phase 2: softmax over each of the 16 rows (16 threads per row) ----
    const int row = tid >> 4;
    const int sub = tid & 15;
    float* __restrict__ srow = sc + row * SEQ;

    float mx = -INFINITY;
    for (int s = sub; s < SEQ; s += 16) mx = fmaxf(mx, srow[s]);
    red[tid] = mx;
    __syncthreads();
    float rmx = red[row * 16];
    #pragma unroll
    for (int j = 1; j < 16; ++j) rmx = fmaxf(rmx, red[row * 16 + j]);
    __syncthreads();

    float sum = 0.f;
    for (int s = sub; s < SEQ; s += 16) {
        const float e = __expf(srow[s] - rmx);
        srow[s] = e;
        sum += e;
    }
    red[tid] = sum;
    __syncthreads();
    float rs = 0.f;
    #pragma unroll
    for (int j = 0; j < 16; ++j) rs += red[row * 16 + j];
    const float inv = 1.0f / rs;
    __syncthreads();

    // normalize in LDS + coalesced float4 writeback of the attention matrix
    float* __restrict__ arow_g = attnp + ((size_t)(h * SEQ + q0 + row)) * SEQ;
    for (int s = sub * 4; s < SEQ; s += 64) {
        float4 p;
        p.x = srow[s + 0] * inv;
        p.y = srow[s + 1] * inv;
        p.z = srow[s + 2] * inv;
        p.w = srow[s + 3] * inv;
        *(float4*)(srow + s) = p;          // keep normalized attn for phase 3
        *(float4*)(arow_g + s) = p;        // stream attn to HBM (256B/halfwave)
    }
    __syncthreads();

    // ---- Phase 3: out = attn @ V  (4 waves, one 16x16 tile each, K=2048) ----
    if (wave < 4) {
        const int wv = __builtin_amdgcn_readfirstlane(wave);  // SGPR-uniform
        const int ncol = wv * 16 + mrow;
        v8f acc = {};

#if FDA_HAS_TDM
        // per-wave private double-buffered V staging (32x16 f32 each)
        float* vst[2] = { sm + VST_BASE + (wv * 2 + 0) * VST_FLOATS,
                          sm + VST_BASE + (wv * 2 + 1) * VST_FLOATS };
        const unsigned off[2] = { (unsigned)(uintptr_t)vst[0],
                                  (unsigned)(uintptr_t)vst[1] };
        const float* __restrict__ vb = v + (size_t)(h * SEQ) * DK + wv * 16;

        tdm_load_v_tile(vb, off[0]);
        for (int kb = 0; kb < SEQ / 32; ++kb) {
            const int cur = kb & 1;
            if (kb + 1 < SEQ / 32) {
                tdm_load_v_tile(vb + (size_t)(kb + 1) * 32 * DK, off[cur ^ 1]);
                __builtin_amdgcn_s_wait_tensorcnt(1);   // oldest DMA done
            } else {
                __builtin_amdgcn_s_wait_tensorcnt(0);
            }
            // A fragment: attn rows from LDS (f32 -> f16)
            const float* __restrict__ sr = sc + mrow * SEQ + kb * 32;
            v16h a2;
            #pragma unroll
            for (int i = 0; i < 8; ++i) {
                a2[i]     = (_Float16)sr[hi * 8 + i];
                a2[8 + i] = (_Float16)sr[16 + hi * 8 + i];
            }
            // B fragment: V columns from the TDM-staged LDS tile
            const float* __restrict__ vt = vst[cur];
            v16h b2;
            #pragma unroll
            for (int i = 0; i < 8; ++i) {
                b2[i]     = (_Float16)vt[(hi * 8 + i) * 16 + mrow];
                b2[8 + i] = (_Float16)vt[(16 + hi * 8 + i) * 16 + mrow];
            }
            acc = __builtin_amdgcn_wmma_f32_16x16x32_f16(
                      false, a2, false, b2, (short)0, acc, false, false);
        }
#else
        // fallback: direct global gather of V columns
        const float* __restrict__ vb = v + (size_t)(h * SEQ) * DK;
        for (int kb = 0; kb < SEQ / 32; ++kb) {
            const int kbase = kb * 32;
            const float* __restrict__ sr = sc + mrow * SEQ + kbase;
            v16h a2;
            #pragma unroll
            for (int i = 0; i < 8; ++i) {
                a2[i]     = (_Float16)sr[hi * 8 + i];
                a2[8 + i] = (_Float16)sr[16 + hi * 8 + i];
            }
            v16h b2;
            #pragma unroll
            for (int i = 0; i < 8; ++i) {
                b2[i]     = (_Float16)vb[(size_t)(kbase + hi * 8 + i) * DK + ncol];
                b2[8 + i] = (_Float16)vb[(size_t)(kbase + 16 + hi * 8 + i) * DK + ncol];
            }
            acc = __builtin_amdgcn_wmma_f32_16x16x32_f16(
                      false, a2, false, b2, (short)0, acc, false, false);
        }
#endif
        #pragma unroll
        for (int r = 0; r < 8; ++r)
            outp[((size_t)(h * SEQ + q0 + r + 8 * hi)) * DK + ncol] = acc[r];
    }
}

// ---------------------------------------------------------------------------
extern "C" void kernel_launch(void* const* d_in, const int* in_sizes, int n_in,
                              void* d_out, int out_size, void* d_ws, size_t ws_size,
                              hipStream_t stream)
{
    const float* q  = (const float*)d_in[0];   // [1,16,2048,64]
    const float* v  = (const float*)d_in[1];   // [1,16,2048,64]
    const float* Wa = (const float*)d_in[2];   // [64,32]
    const float* ba = (const float*)d_in[3];   // [32]
    const float* Wb = (const float*)d_in[4];   // [64,64]
    const float* bb = (const float*)d_in[5];   // [64]
    // d_in[6] = len_q (== 2048, full sequence)

    float* outp  = (float*)d_out;                                // [16,2048,64]
    float* attnp = (float*)d_out + (size_t)NHEAD * SEQ * DK;     // [16,2048,2048]

    _Float16* Af = (_Float16*)d_ws;                              // [16,2048,32]
    _Float16* Cf = Af + (size_t)NHEAD * SEQ * 32;                // [16,2048,32]

    // Kernel 1: projections (H*S = 32768 rows, 1 thread/row)
    fda_precompute<<<(NHEAD * SEQ) / 256, 256, 0, stream>>>(q, Wa, ba, Wb, bb, Af, Cf);

    // Kernel 2: fused attention, ~145KB dynamic LDS per block
    const size_t lds = (size_t)LDS_FLOATS * sizeof(float);
    fda_attention<<<NHEAD * (SEQ / 16), 256, lds, stream>>>(Af, Cf, v, outp, attnp);
}
